// MultiHeadAttention_22840636080592
// MI455X (gfx1250) — matrix operand order repaired and checked by reading, not compiled, not run on hardware
//
#include <hip/hip_runtime.h>
#include <hip/hip_bf16.h>
#include <cstdint>
#include <cstddef>

// ---------------------------------------------------------------------------
// Multi-head attention for MI455X (gfx1250), wave32 + V_WMMA_F32_16X16X32_F16
// Pipeline:
//   1) fp32 -> f16 conversion (weights once-per-call, activations via reused X)
//   2) Q/K/V projection GEMMs (f16 tiles, async global->LDS, double-buffered,
//      BK=64 -> 16 WMMAs per barrier)
//   3) flash attention: K via async DMA, V via software-pipelined transposed
//      staging (so every WMMA operand read is ds_load_b128), online softmax
//   4) output projection GEMM (f16 -> fp32 d_out)
// ---------------------------------------------------------------------------

namespace {
constexpr int kEmbed = 1024;
constexpr int kHeads = 16;
constexpr int kHdim  = 64;
constexpr int kBatch = 4;
constexpr int kSeq   = 2048;
constexpr int kM     = kBatch * kSeq;   // 8192 rows for the big GEMMs
constexpr float kNeg = -1.0e20f;
}

typedef __attribute__((ext_vector_type(16))) _Float16 v16h;
typedef __attribute__((ext_vector_type(8)))  float    v8f;

union FragH { v16h h; unsigned int u[8]; };
union PairH { _Float16 h[2]; unsigned int u; };

// --------------------------- async copy support ----------------------------
#if defined(__has_builtin)
#  if __has_builtin(__builtin_amdgcn_global_load_async_to_lds_b128) && \
      __has_builtin(__builtin_amdgcn_s_wait_asynccnt)
#    define USE_ASYNC_COPY 1
#  endif
#endif
#ifndef USE_ASYNC_COPY
#  define USE_ASYNC_COPY 0
#endif

#if USE_ASYNC_COPY
// Builtin signature (from hipcc diagnostic): (int4 AS1*, int4 AS3*, Imm, Imm)
typedef int v4i __attribute__((ext_vector_type(4)));
typedef v4i __attribute__((address_space(1))) gv4i;   // global
typedef v4i __attribute__((address_space(3))) lv4i;   // LDS
#endif

// 16-byte global -> LDS tile copy. Async (ASYNCcnt) when available; otherwise
// a plain register round-trip.
__device__ __forceinline__ void lds_copy16(void* lds_dst, const void* gsrc) {
#if USE_ASYNC_COPY
  __builtin_amdgcn_global_load_async_to_lds_b128(
      (gv4i*)gsrc, (lv4i*)lds_dst, /*offset=*/0, /*cpol=*/0);
#else
  *(uint4*)lds_dst = *(const uint4*)gsrc;
#endif
}

__device__ __forceinline__ void async_wait0() {
#if USE_ASYNC_COPY
  __builtin_amdgcn_s_wait_asynccnt(0);
#endif
}

// ------------------------------- WMMA core ---------------------------------
__device__ __forceinline__ v8f wmma32(v16h a, v16h b, v8f c) {
  // D = A(16x32 f16) * B(32x16 f16) + C(16x16 f32)
  return __builtin_amdgcn_wmma_f32_16x16x32_f16(
      /*neg_a=*/false, a, /*neg_b=*/false, b,
      /*c_mod=*/(short)0, c, /*reuse_a=*/false, /*reuse_b=*/false);
}

// A-fragment (16x32, row-major in LDS: lds[(m0+lane15)*ldk + kk + k]):
// lane L: m = m0 + L%16, half = L/16
// dword r    : K = kk + 16*half + 2r      (16 contiguous halves -> 2x ds_load_b128)
__device__ __forceinline__ v16h load_a_frag(const _Float16* lds, int ldk,
                                            int m0, int kk, int lane15, int half) {
  FragH f;
  const _Float16* row = lds + (size_t)(m0 + lane15) * ldk + kk;
#pragma unroll
  for (int r = 0; r < 8; ++r) {
    const int kb = 16 * half + 2 * r;
    f.u[r] = *(const unsigned int*)(row + kb);
  }
  return f.h;
}

// B-fragment where the logical B(K x N) is stored transposed in LDS as
// src[n][k] (W^T, K^T and transposed-V cases): lane L: n = n0 + L%16;
// dword r: K = kk + 16*half + 2r  -> 16 contiguous halves -> 2x ds_load_b128.
__device__ __forceinline__ v16h load_bT_frag(const _Float16* lds, int ldk,
                                             int n0, int kk, int lane15, int half) {
  FragH f;
  const _Float16* row = lds + (size_t)(n0 + lane15) * ldk + kk + 16 * half;
#pragma unroll
  for (int r = 0; r < 8; ++r) f.u[r] = *(const unsigned int*)(row + 2 * r);
  return f.h;
}

// ---------------------------------------------------------------------------
// fp32 -> f16 elementwise. n must be a multiple of 4 and at most 4*256*gridDim.
// ---------------------------------------------------------------------------
__global__ __launch_bounds__(256)
void f32_to_f16_kernel(const float* __restrict__ src, _Float16* __restrict__ dst, int n) {
  int i = (blockIdx.x * 256 + threadIdx.x) * 4;
  if (i + 3 < n) {
    float4 v = *(const float4*)(src + i);
    PairH p0, p1;
    p0.h[0] = (_Float16)v.x; p0.h[1] = (_Float16)v.y;
    p1.h[0] = (_Float16)v.z; p1.h[1] = (_Float16)v.w;
    *(unsigned int*)(dst + i)     = p0.u;
    *(unsigned int*)(dst + i + 2) = p1.u;
  }
}

// ---------------------------------------------------------------------------
// GEMM: Y = alpha * A(MxK, f16) @ W(NxK, f16)^T
//   OUT_F32 : Y fp32 (output projection) else f16 (Q/K/V projections)
// Block tile 128x128x64, 8 waves, wave tile 32x64 (2x4 WMMA tiles, 2 k-steps
// per staged tile -> 16 WMMAs per barrier). Async double-buffered LDS.
// ---------------------------------------------------------------------------
template <bool OUT_F32>
__global__ __launch_bounds__(256)
void gemm_f16_xwT_kernel(const _Float16* __restrict__ A,
                         const _Float16* __restrict__ W,
                         void* __restrict__ Yptr,
                         int M, int N, int K, float alpha) {
  constexpr int BM = 128, BN = 128, BK = 64;
  __shared__ __align__(16) _Float16 As[2][BM][BK];   // 2 x 16 KB
  __shared__ __align__(16) _Float16 Bs[2][BN][BK];   // 2 x 16 KB

  const int tid = threadIdx.x;
  const int wave = tid >> 5, lane = tid & 31;
  const int lane15 = lane & 15, half = lane >> 4;
  const int bm = blockIdx.y * BM, bn = blockIdx.x * BN;
  const int wm = (wave >> 1) * 32;       // 0,32,64,96
  const int wn = (wave & 1) * 64;        // 0,64

  // staging coords: 1024 uint4 per 128x64 tile -> 4 per thread
  const int srow = tid >> 3;             // 0..31 (+32 per pass)
  const int scol = (tid & 7) * 8;

  auto stage = [&](int buf, int k0) {
#pragma unroll
    for (int i = 0; i < 4; ++i) {
      const int row = srow + i * 32;
      lds_copy16(&As[buf][row][scol], A + (size_t)(bm + row) * K + k0 + scol);
      lds_copy16(&Bs[buf][row][scol], W + (size_t)(bn + row) * K + k0 + scol);
    }
  };

  v8f acc[2][4];
#pragma unroll
  for (int tm = 0; tm < 2; ++tm)
#pragma unroll
    for (int tn = 0; tn < 4; ++tn) acc[tm][tn] = {};

  stage(0, 0);
  int buf = 0;
  for (int k0 = 0; k0 < K; k0 += BK) {
    async_wait0();        // our own async stores to LDS are complete
    __syncthreads();      // everyone's are complete; prev compute finished
    if (k0 + BK < K) stage(buf ^ 1, k0 + BK);   // overlap next tile's flight

#pragma unroll
    for (int kk = 0; kk < BK; kk += 32) {
      v16h a0 = load_a_frag(&As[buf][0][0], BK, wm + 0,  kk, lane15, half);
      v16h a1 = load_a_frag(&As[buf][0][0], BK, wm + 16, kk, lane15, half);
#pragma unroll
      for (int tn = 0; tn < 4; ++tn) {
        v16h bf = load_bT_frag(&Bs[buf][0][0], BK, wn + 16 * tn, kk, lane15, half);
        acc[0][tn] = wmma32(a0, bf, acc[0][tn]);
        acc[1][tn] = wmma32(a1, bf, acc[1][tn]);
      }
    }
    buf ^= 1;
  }

  // ---- epilogue ----
#pragma unroll
  for (int tm = 0; tm < 2; ++tm) {
#pragma unroll
    for (int tn = 0; tn < 4; ++tn) {
      const int ncol = bn + wn + tn * 16 + lane15;
#pragma unroll
      for (int r = 0; r < 8; ++r) {
        const int mrow = bm + wm + tm * 16 + r + 8 * half;
        const float v = alpha * acc[tm][tn][r];
        if constexpr (OUT_F32)
          ((float*)Yptr)[(size_t)mrow * N + ncol] = v;
        else
          ((_Float16*)Yptr)[(size_t)mrow * N + ncol] = (_Float16)v;
      }
    }
  }
}

// ---------------------------------------------------------------------------
// Flash attention: per block = one (b, h) pair and 128 query rows.
// Q/K/V f16 [B, S, H*Dh]; scores/softmax fp32; output f16 [B, S, H*Dh].
// Wave w owns query rows [16w, 16w+16). K tiles via async DMA (double-
// buffered); V tiles software-pipelined through registers and stored
// TRANSPOSED (Vt[d][k]) so P@V B-fragments are contiguous ds_load_b128.
// ---------------------------------------------------------------------------
__global__ __launch_bounds__(256)
void flash_attn_kernel(const _Float16* __restrict__ Qh,
                       const _Float16* __restrict__ Kh,
                       const _Float16* __restrict__ Vh,
                       const int* __restrict__ mask,
                       const int* __restrict__ padding,
                       _Float16* __restrict__ Ah) {
  constexpr int Br = 128, Bc = 64, D = kHdim;
  __shared__ __align__(16) _Float16 Qs[Br][D];          // 16 KB
  __shared__ __align__(16) _Float16 Ks[2][Bc][D];       // 2 x 8 KB
  __shared__ __align__(16) _Float16 Vt[2][D][Bc];       // 2 x 8 KB (transposed)
  __shared__ __align__(16) _Float16 Ps[8][16][Bc];      // 16 KB (per-wave P)

  const int bh = blockIdx.y;
  const int b = bh / kHeads, h = bh % kHeads;
  const int qbase = blockIdx.x * Br;
  const int tid = threadIdx.x;
  const int wave = tid >> 5, lane = tid & 31;
  const int lane15 = lane & 15, half = lane >> 4;

  const _Float16* Qg  = Qh + ((size_t)b * kSeq) * kEmbed + h * D;
  const _Float16* Kg0 = Kh + ((size_t)b * kSeq) * kEmbed + h * D;
  const _Float16* Vg0 = Vh + ((size_t)b * kSeq) * kEmbed + h * D;

  // staging coords (rows of 64 f16 = 8 uint4)
  const int qrow0 = tid >> 3;            // 0..31 (+32 per pass)
  const int qcol  = (tid & 7) * 8;

  auto stageK = [&](int buf, int j) {
#pragma unroll
    for (int i = 0; i < 2; ++i) {
      const int row = qrow0 + i * 32;    // 0..63
      lds_copy16(&Ks[buf][row][qcol],
                 Kg0 + (size_t)(j + row) * kEmbed + qcol);
    }
  };

  // V software pipeline: issue global loads for tile j, commit transposed
  // one iteration later (just before the barrier that publishes the buffer).
  uint4 vreg0, vreg1;
  auto issueV = [&](int j) {
    vreg0 = *(const uint4*)(Vg0 + (size_t)(j + qrow0) * kEmbed + qcol);
    vreg1 = *(const uint4*)(Vg0 + (size_t)(j + qrow0 + 32) * kEmbed + qcol);
  };
  auto commitV = [&](int buf) {
    union { uint4 q; _Float16 e[8]; } u0, u1;
    u0.q = vreg0; u1.q = vreg1;
#pragma unroll
    for (int e = 0; e < 8; ++e) {
      Vt[buf][qcol + e][qrow0]      = u0.e[e];
      Vt[buf][qcol + e][qrow0 + 32] = u1.e[e];
    }
  };

  // prologue: Q tile (async) + first K tile (async) + first V tile (regs)
#pragma unroll
  for (int i = 0; i < 4; ++i) {
    const int row = qrow0 + i * 32;      // 0..127
    lds_copy16(&Qs[row][qcol], Qg + (size_t)(qbase + row) * kEmbed + qcol);
  }
  stageK(0, 0);
  issueV(0);

  float m_i[8], l_i[8];
  v8f acc_o[4];
#pragma unroll
  for (int r = 0; r < 8; ++r) { m_i[r] = -3.0e38f; l_i[r] = 0.0f; }
#pragma unroll
  for (int tn = 0; tn < 4; ++tn) acc_o[tn] = {};

  int buf = 0;
  for (int j = 0; j < kSeq; j += Bc) {
    async_wait0();        // K[buf] async complete (per-wave)
    commitV(buf);         // publish transposed V[buf] from registers
    __syncthreads();      // all waves' K DMA + V stores visible
    if (j + Bc < kSeq) {
      stageK(buf ^ 1, j + Bc);          // next K tile in flight
      issueV(j + Bc);                   // next V tile into registers
    }

    // ---- S = Q @ K^T  (16 x 64 per wave, fp32 acc) ----
    v8f s[4];
#pragma unroll
    for (int tn = 0; tn < 4; ++tn) s[tn] = {};
#pragma unroll
    for (int kk = 0; kk < D; kk += 32) {
      v16h a = load_a_frag(&Qs[0][0], D, wave * 16, kk, lane15, half);
#pragma unroll
      for (int tn = 0; tn < 4; ++tn) {
        v16h bf = load_bT_frag(&Ks[buf][0][0], D, tn * 16, kk, lane15, half);
        s[tn] = wmma32(a, bf, s[tn]);
      }
    }

    // ---- mask + online softmax ----
    int pd[4];
#pragma unroll
    for (int tn = 0; tn < 4; ++tn)
      pd[tn] = padding[(size_t)b * kSeq + j + tn * 16 + lane15];

#pragma unroll
    for (int r = 0; r < 8; ++r) {
      const int qrow = qbase + wave * 16 + r + 8 * half;
      const size_t mrow = (size_t)qrow * kSeq + j;
      float rm = -3.0e38f;
#pragma unroll
      for (int tn = 0; tn < 4; ++tn) {
        const int mk = mask[mrow + tn * 16 + lane15];
        float sv = (mk != 0 && pd[tn] != 0) ? s[tn][r] : kNeg;
        s[tn][r] = sv;
        rm = fmaxf(rm, sv);
      }
      // 16-lane butterfly (xor masks 1..8 stay within each half-wave group)
#pragma unroll
      for (int off = 1; off < 16; off <<= 1)
        rm = fmaxf(rm, __shfl_xor(rm, off));

      const float mnew = fmaxf(m_i[r], rm);
      const float sc = __expf(m_i[r] - mnew);
      l_i[r] *= sc;
      m_i[r] = mnew;
#pragma unroll
      for (int tn = 0; tn < 4; ++tn) acc_o[tn][r] *= sc;

      float rs = 0.0f;
#pragma unroll
      for (int tn = 0; tn < 4; ++tn) {
        const float p = __expf(s[tn][r] - mnew);
        rs += p;
        Ps[wave][r + 8 * half][tn * 16 + lane15] = (_Float16)p;
      }
#pragma unroll
      for (int off = 1; off < 16; off <<= 1)
        rs += __shfl_xor(rs, off);
      l_i[r] += rs;
    }

    // ---- O += P @ V  (Vt is [d][k]: contiguous B-fragment reads) ----
#pragma unroll
    for (int kk = 0; kk < Bc; kk += 32) {
      v16h a = load_a_frag(&Ps[wave][0][0], Bc, 0, kk, lane15, half);
#pragma unroll
      for (int tn = 0; tn < 4; ++tn) {
        v16h bf = load_bT_frag(&Vt[buf][0][0], Bc, tn * 16, kk, lane15, half);
        acc_o[tn] = wmma32(a, bf, acc_o[tn]);
      }
    }
    buf ^= 1;
  }

  // ---- normalize and store (f16 [B, S, H*Dh]) ----
  _Float16* Ag = Ah + ((size_t)b * kSeq) * kEmbed + h * D;
#pragma unroll
  for (int r = 0; r < 8; ++r) {
    const int qrow = qbase + wave * 16 + r + 8 * half;
    const float inv = 1.0f / fmaxf(l_i[r], 1.0e-20f);
#pragma unroll
    for (int tn = 0; tn < 4; ++tn) {
      Ag[(size_t)qrow * kEmbed + tn * 16 + lane15] =
          (_Float16)(acc_o[tn][r] * inv);
    }
  }
}

// ---------------------------------------------------------------------------
// Host launcher. Inputs: query,key,value,mask,padding,Wq,Wk,Wv,Wo (fp32/int32)
// Output: fp32 [B, S, EMBED]. Workspace (72 MB): 4 f16 weights, one reused
// 16 MB f16 activation buffer X0, and f16 Q/K/V.
// ---------------------------------------------------------------------------
extern "C" void kernel_launch(void* const* d_in, const int* in_sizes, int n_in,
                              void* d_out, int out_size, void* d_ws, size_t ws_size,
                              hipStream_t stream) {
  (void)in_sizes; (void)n_in; (void)out_size; (void)ws_size;

  const float* q    = (const float*)d_in[0];
  const float* k    = (const float*)d_in[1];
  const float* v    = (const float*)d_in[2];
  const int*   mask = (const int*)d_in[3];
  const int*   pad  = (const int*)d_in[4];
  const float* Wq   = (const float*)d_in[5];
  const float* Wk   = (const float*)d_in[6];
  const float* Wv   = (const float*)d_in[7];
  const float* Wo   = (const float*)d_in[8];
  float* out = (float*)d_out;

  _Float16* Wqh = (_Float16*)d_ws;
  _Float16* Wkh = Wqh + (size_t)kEmbed * kEmbed;
  _Float16* Wvh = Wkh + (size_t)kEmbed * kEmbed;
  _Float16* Woh = Wvh + (size_t)kEmbed * kEmbed;
  _Float16* X0  = Woh + (size_t)kEmbed * kEmbed;   // reused: Xq, Xk, Xv, attn-out
  _Float16* Qh  = X0  + (size_t)kM * kEmbed;
  _Float16* Kh  = Qh  + (size_t)kM * kEmbed;
  _Float16* Vh  = Kh  + (size_t)kM * kEmbed;

  const int wn = kEmbed * kEmbed;                  // 1M elems per weight
  const int xn = kM * kEmbed;                      // 8M elems per activation
  const dim3 cvt_w(wn / (256 * 4));
  const dim3 cvt_x(xn / (256 * 4));
  f32_to_f16_kernel<<<cvt_w, 256, 0, stream>>>(Wq, Wqh, wn);
  f32_to_f16_kernel<<<cvt_w, 256, 0, stream>>>(Wk, Wkh, wn);
  f32_to_f16_kernel<<<cvt_w, 256, 0, stream>>>(Wv, Wvh, wn);
  f32_to_f16_kernel<<<cvt_w, 256, 0, stream>>>(Wo, Woh, wn);

  // qs = (x @ Wq^T) / E^0.25, ks likewise; vs unscaled
  const float alpha_qk = 0.17677669529663687f;     // 1024^-0.25
  const dim3 gemm_grid(kEmbed / 128, kM / 128);    // (8, 64)

  f32_to_f16_kernel<<<cvt_x, 256, 0, stream>>>(q, X0, xn);
  gemm_f16_xwT_kernel<false><<<gemm_grid, 256, 0, stream>>>(
      X0, Wqh, Qh, kM, kEmbed, kEmbed, alpha_qk);

  f32_to_f16_kernel<<<cvt_x, 256, 0, stream>>>(k, X0, xn);
  gemm_f16_xwT_kernel<false><<<gemm_grid, 256, 0, stream>>>(
      X0, Wkh, Kh, kM, kEmbed, kEmbed, alpha_qk);

  f32_to_f16_kernel<<<cvt_x, 256, 0, stream>>>(v, X0, xn);
  gemm_f16_xwT_kernel<false><<<gemm_grid, 256, 0, stream>>>(
      X0, Wvh, Vh, kM, kEmbed, kEmbed, 1.0f);

  // attention output reuses X0 (dead after the V projection)
  const dim3 fa_grid(kSeq / 128, kBatch * kHeads); // (16, 64)
  flash_attn_kernel<<<fa_grid, 256, 0, stream>>>(Qh, Kh, Vh, mask, pad, X0);

  gemm_f16_xwT_kernel<true><<<gemm_grid, 256, 0, stream>>>(
      X0, Woh, out, kM, kEmbed, kEmbed, 1.0f);
}
